// RuleClassifierSNN_3272765079885
// MI455X (gfx1250) — matrix operand order repaired and checked by reading, not compiled, not run on hardware
//
#include <hip/hip_runtime.h>

// ---------------------------------------------------------------------------
// RuleClassifierSNN forward for MI455X (gfx1250, wave32, WMMA).
//
// Forward-only algebra: the scan carries the POST-RESET membrane (v < 1.0
// everywhere), so surrogate_spike(v) == 0 and the W_rec recurrent term is
// identically zero in the forward pass => W_rec (d_in[6]) unused.
//
// Per workgroup (256 threads = 8 wave32, 16 batch rows):
//   * proj[v][h] = emb @ W_in^T + b_in as a 128-row f16 LUT in LDS
//     (row stride padded to 136 halfs = 272B to kill LDS bank conflicts).
//   * Token ids staged as u8[t][16] (4KB, conflict-free lane reads).
//   * Wave w owns output columns [16w,16w+16); its W_lif B-fragments
//     (2 layers x 4 K-chunks x v16h) are loaded ONCE into registers.
//   * T-loop processes TWO timesteps per barrier: layer0(t), layer0(t+1),
//     prefetch ids+A-frags for t+2/t+3, ONE barrier, layer1(t), layer1(t+1).
//     Spikes rotate through 4 padded LDS buffers (no WAR barrier needed).
//   * Epilogue: spike_acc/256 -> 16x12 classifier (reuses spike LDS).
// ---------------------------------------------------------------------------

typedef __attribute__((ext_vector_type(16))) _Float16 v16h;
typedef __attribute__((ext_vector_type(8)))  _Float16 v8h;
typedef __attribute__((ext_vector_type(8)))  float    v8f;

#define SNN_B    2048
#define SNN_T    256
#define SNN_H    128
#define SNN_E    64
#define SNN_C    12
#define SNN_BETA 0.8f

#define PSTR     136                    // padded row stride (halfs) for proj/s0
#define S0_HALFS (16 * PSTR)            // 2176 halfs per spike buffer

// LDS layout (total 56320 B)
#define OFF_PROJ 0                      // 128*136*2 = 34816 B
#define OFF_IDS  34816                  // u8[256][16] = 4096 B
#define OFF_S0   38912                  // 4 x 16*136*2 = 17408 B (reused as f32 acc)
#define SMEM_BYTES 56320

__device__ __forceinline__ v16h ldA(const _Float16* p) {
    v8h lo = *(const v8h*)(p);
    v8h hi = *(const v8h*)(p + 16);
    return __builtin_shufflevector(lo, hi, 0,1,2,3,4,5,6,7,8,9,10,11,12,13,14,15);
}

__global__ __launch_bounds__(256)
void snn_forward_kernel(const int*   __restrict__ ids_g,    // [B,T]
                        const float* __restrict__ emb_g,    // [V,E]
                        const float* __restrict__ Win_g,    // [H,E]
                        const float* __restrict__ bin_g,    // [H]
                        const float* __restrict__ Wlif_g,   // [2,H,H]
                        const float* __restrict__ blif_g,   // [2,H]
                        const float* __restrict__ Wcls_g,   // [C,H]
                        const float* __restrict__ bcls_g,   // [C]
                        float*       __restrict__ out_g)    // [B,C]
{
    __shared__ __align__(16) char smem[SMEM_BYTES];
    _Float16*      proj  = (_Float16*)(smem + OFF_PROJ);
    unsigned char* idsb  = (unsigned char*)(smem + OFF_IDS);
    _Float16*      winh  = (_Float16*)(smem + OFF_IDS);   // transient (16KB spans ids+s0)
    _Float16*      s0b   = (_Float16*)(smem + OFF_S0);
    float*         accb  = (float*)   (smem + OFF_S0);    // epilogue reuse

    const int tid  = threadIdx.x;
    const int lane = tid & 31;
    const int wv   = tid >> 5;       // wave 0..7 -> output columns [16wv,16wv+16)
    const int n0   = wv << 4;
    const int nl   = lane & 15;      // column within tile / A-frag row
    const int hi   = lane >> 4;      // half-wave K sub-range select
    const int aoff = hi << 3;        // A-frag K sub-offset (halfs)
    const int b0   = blockIdx.x << 4;

    // ---- Stage W_in as f16 (transient region) ----
    #pragma unroll
    for (int j = 0; j < 32; ++j) {
        int i = tid * 32 + j;
        winh[i] = (_Float16)Win_g[i];
    }
    __syncthreads();

    // ---- proj LUT: proj[v][h] = sum_e emb[v][e]*W_in[h][e] + b_in[h] ----
    {
        const int idv   = tid >> 1;
        const int hbase = (tid & 1) << 6;
        float embr[SNN_E];
        #pragma unroll
        for (int e = 0; e < SNN_E; ++e) embr[e] = emb_g[idv * SNN_E + e];
        for (int hh = 0; hh < 64; ++hh) {
            const int h = hbase + hh;
            float s = bin_g[h];
            const _Float16* wr = winh + h * SNN_E;
            #pragma unroll
            for (int e = 0; e < SNN_E; ++e) s += embr[e] * (float)wr[e];
            proj[idv * PSTR + h] = (_Float16)s;
        }
    }
    __syncthreads();   // proj done; winh region now free

    // ---- Stage token ids as u8, transposed: idsb[t*16 + row] ----
    {
        const int row   = tid >> 4;
        const int tbase = (tid & 15) << 4;
        #pragma unroll
        for (int j = 0; j < 16; ++j) {
            int v = ids_g[(b0 + row) * SNN_T + tbase + j];
            idsb[(tbase + j) * 16 + row] = (unsigned char)v;
        }
    }

    // ---- W_lif B-fragments -> registers (held for all T) ----
    v16h wf[2][4];
    #pragma unroll
    for (int l = 0; l < 2; ++l) {
        #pragma unroll
        for (int kc = 0; kc < 4; ++kc) {
            const float* wp = Wlif_g + ((l * SNN_H + n0 + nl) * SNN_H) + kc * 32 + hi * 8;
            v16h w;
            #pragma unroll
            for (int j = 0; j < 16; ++j) w[j] = (_Float16)wp[j < 8 ? j : j + 8];
            wf[l][kc] = w;
        }
    }
    const float bl0 = blif_g[n0 + nl];
    const float bl1 = blif_g[SNN_H + n0 + nl];
    __syncthreads();

    // ---- Pipelined prologue: A-frags for t=0, id for t=1 ----
    v16h a0A[4];
    {
        const _Float16* pr = proj + (int)idsb[0 * 16 + nl] * PSTR;
        #pragma unroll
        for (int kc = 0; kc < 4; ++kc) a0A[kc] = ldA(pr + kc * 32 + aoff);
    }
    int idB = idsb[1 * 16 + nl];

    v8f v0 = {}, v1 = {}, accum = {};

    // ---- Main loop: 2 timesteps per barrier, 4 rotating spike buffers ----
    #pragma unroll 2
    for (int k = 0; k < SNN_T / 2; ++k) {
        const int t = k << 1;
        _Float16* sbA = s0b + ((k & 1) ? 2 * S0_HALFS : 0);
        _Float16* sbB = sbA + S0_HALFS;

        // ===== layer 0, step A =====
        v8f c0 = {bl0, bl0, bl0, bl0, bl0, bl0, bl0, bl0};
        #pragma unroll
        for (int kc = 0; kc < 4; ++kc)
            c0 = __builtin_amdgcn_wmma_f32_16x16x32_f16(
                     false, a0A[kc], false, wf[0][kc], (short)0, c0, false, false);

        // load step-B A-frags (overlaps step-A WMMAs)
        v16h a0B[4];
        {
            const _Float16* pr = proj + idB * PSTR;
            #pragma unroll
            for (int kc = 0; kc < 4; ++kc) a0B[kc] = ldA(pr + kc * 32 + aoff);
        }

        #pragma unroll
        for (int r = 0; r < 8; ++r) {
            float vn = SNN_BETA * v0[r] + (1.0f - SNN_BETA) * c0[r];
            bool sp = (vn >= 1.0f);
            sbA[(r + (hi << 3)) * PSTR + n0 + nl] = sp ? (_Float16)1.0f : (_Float16)0.0f;
            v0[r] = sp ? 0.0f : vn;
        }

        // ===== layer 0, step B =====
        v8f c0b = {bl0, bl0, bl0, bl0, bl0, bl0, bl0, bl0};
        #pragma unroll
        for (int kc = 0; kc < 4; ++kc)
            c0b = __builtin_amdgcn_wmma_f32_16x16x32_f16(
                      false, a0B[kc], false, wf[0][kc], (short)0, c0b, false, false);

        // prefetch next pair: id(t+2) -> A-frags, id(t+3) -> register
        const int tn = (t + 2) & (SNN_T - 1);
        {
            const _Float16* pr = proj + (int)idsb[tn * 16 + nl] * PSTR;
            #pragma unroll
            for (int kc = 0; kc < 4; ++kc) a0A[kc] = ldA(pr + kc * 32 + aoff);
        }
        idB = idsb[((t + 3) & (SNN_T - 1)) * 16 + nl];

        #pragma unroll
        for (int r = 0; r < 8; ++r) {
            float vn = SNN_BETA * v0[r] + (1.0f - SNN_BETA) * c0b[r];
            bool sp = (vn >= 1.0f);
            sbB[(r + (hi << 3)) * PSTR + n0 + nl] = sp ? (_Float16)1.0f : (_Float16)0.0f;
            v0[r] = sp ? 0.0f : vn;
        }

        __syncthreads();   // the ONLY barrier per 2 timesteps

        // ===== layer 1, step A =====
        {
            v16h aA[4];
            #pragma unroll
            for (int kc = 0; kc < 4; ++kc)
                aA[kc] = ldA(sbA + nl * PSTR + kc * 32 + aoff);
            v8f c1 = {bl1, bl1, bl1, bl1, bl1, bl1, bl1, bl1};
            #pragma unroll
            for (int kc = 0; kc < 4; ++kc)
                c1 = __builtin_amdgcn_wmma_f32_16x16x32_f16(
                         false, aA[kc], false, wf[1][kc], (short)0, c1, false, false);
            #pragma unroll
            for (int r = 0; r < 8; ++r) {
                float vn = SNN_BETA * v1[r] + (1.0f - SNN_BETA) * c1[r];
                bool sp = (vn >= 1.0f);
                v1[r] = sp ? 0.0f : vn;
                accum[r] += sp ? 1.0f : 0.0f;
            }
        }
        // ===== layer 1, step B =====
        {
            v16h aB[4];
            #pragma unroll
            for (int kc = 0; kc < 4; ++kc)
                aB[kc] = ldA(sbB + nl * PSTR + kc * 32 + aoff);
            v8f c1 = {bl1, bl1, bl1, bl1, bl1, bl1, bl1, bl1};
            #pragma unroll
            for (int kc = 0; kc < 4; ++kc)
                c1 = __builtin_amdgcn_wmma_f32_16x16x32_f16(
                         false, aB[kc], false, wf[1][kc], (short)0, c1, false, false);
            #pragma unroll
            for (int r = 0; r < 8; ++r) {
                float vn = SNN_BETA * v1[r] + (1.0f - SNN_BETA) * c1[r];
                bool sp = (vn >= 1.0f);
                v1[r] = sp ? 0.0f : vn;
                accum[r] += sp ? 1.0f : 0.0f;
            }
        }
        // no trailing barrier: next pair writes the other buffer set (WAR-safe)
    }

    // ---- Epilogue: spike_acc/T (reuse spike LDS as f32), 16x12 classifier ----
    __syncthreads();   // all spike-buffer reads done before region reuse
    #pragma unroll
    for (int r = 0; r < 8; ++r)
        accb[(r + (hi << 3)) * SNN_H + n0 + nl] = accum[r] * (1.0f / (float)SNN_T);
    __syncthreads();

    if (tid < 16 * SNN_C) {
        const int row = tid / SNN_C;
        const int c   = tid % SNN_C;
        float s = bcls_g[c];
        const float* wr = Wcls_g + c * SNN_H;
        const float* ar = accb + row * SNN_H;
        #pragma unroll 4
        for (int h = 0; h < SNN_H; ++h) s += ar[h] * wr[h];
        out_g[(b0 + row) * SNN_C + c] = s;
    }
}

extern "C" void kernel_launch(void* const* d_in, const int* in_sizes, int n_in,
                              void* d_out, int out_size, void* d_ws, size_t ws_size,
                              hipStream_t stream) {
    const int*   ids  = (const int*)  d_in[0];   // input_ids [2048,256] i32
    const float* emb  = (const float*)d_in[1];   // [128,64]
    const float* Win  = (const float*)d_in[2];   // [128,64]
    const float* bin  = (const float*)d_in[3];   // [128]
    const float* Wlif = (const float*)d_in[4];   // [2,128,128]
    const float* blif = (const float*)d_in[5];   // [2,128]
    // d_in[6] = W_rec: provably unused in the forward pass (spike(v_post_reset)==0)
    const float* Wcls = (const float*)d_in[7];   // [12,128]
    const float* bcls = (const float*)d_in[8];   // [12]
    float* out = (float*)d_out;                  // [2048,12]

    dim3 grid(SNN_B / 16);   // 128 workgroups, 16 batch rows each
    dim3 block(256);         // 8 wave32 per workgroup
    snn_forward_kernel<<<grid, block, 0, stream>>>(ids, emb, Win, bin,
                                                   Wlif, blif, Wcls, bcls, out);
}
